// PairLoss_50483045597572
// MI455X (gfx1250) — matrix operand order repaired and checked by reading, not compiled, not run on hardware
//
#include <hip/hip_runtime.h>

// ---------------------------------------------------------------------------
// PairLoss for MI455X (gfx1250, wave32, WMMA)
//   loss = mean_valid( max(0, MARGIN - hardest_negative_distance) )
// Gram matrix E*E^T via v_wmma_f32_16x16x32_f16 (f16 in, f32 acc).
// Staging uses GLOBAL_LOAD_ASYNC_TO_LDS_B128 when the toolchain exposes it.
// ---------------------------------------------------------------------------

typedef __attribute__((ext_vector_type(16))) _Float16 v16h;
typedef __attribute__((ext_vector_type(8)))  float    v8f;
typedef int v4i_ __attribute__((vector_size(16)));   // builtin's expected pointee

#define BATCH   8
#define CCH     256
#define HWSZ    (152 * 152)
#define KPTS    1024
#define MARGINF 10.0f
#define BIGF    1e30f

#define TILE    64
#define CHUNK   128
#define LDW     (CHUNK + 8)   // LDS row stride in halves: 68 dwords -> 4 mod 64 banks

#if defined(__has_builtin)
#  if __has_builtin(__builtin_amdgcn_global_load_async_to_lds_b128)
#    define HAVE_ASYNC_LDS 1
#  else
#    define HAVE_ASYNC_LDS 0
#  endif
#else
#  define HAVE_ASYNC_LDS 0
#endif

#define ASG __attribute__((address_space(1)))
#define ASL __attribute__((address_space(3)))

__device__ __forceinline__ void wait_async_zero() {
#if defined(__has_builtin)
#  if __has_builtin(__builtin_amdgcn_s_wait_asynccnt)
    __builtin_amdgcn_s_wait_asynccnt(0);
    return;
#  endif
#endif
    asm volatile("s_wait_asynccnt 0" ::: "memory");
}

union FragH { v16h v; uint4 q[2]; };

// ---------------------------------------------------------------------------
// Kernel 1: gather + L2-normalize one (b,k) row per block; emit f16 row,
// per-row squared norm (after scaling), and init the atomic-min buffer.
// ---------------------------------------------------------------------------
__global__ __launch_bounds__(256) void gather_norm(
    const float* __restrict__ out_id, const int* __restrict__ ind,
    const float* __restrict__ emb_scale, _Float16* __restrict__ embh,
    float* __restrict__ sq, unsigned* __restrict__ negmin)
{
    const int bk = blockIdx.x;          // 0 .. B*K-1
    const int b  = bk / KPTS;
    const int c  = threadIdx.x;         // channel
    const int pos = ind[bk];

    float v = out_id[((size_t)b * CCH + c) * HWSZ + pos];

    // sum of squares across the 256-thread block (8 waves)
    float ss = v * v;
    #pragma unroll
    for (int s = 16; s >= 1; s >>= 1) ss += __shfl_xor(ss, s, 32);
    __shared__ float wss[8];
    const int w = threadIdx.x >> 5, l = threadIdx.x & 31;
    if (l == 0) wss[w] = ss;
    __syncthreads();
    if (threadIdx.x == 0) {
        float t = 0.f;
        #pragma unroll
        for (int i = 0; i < 8; ++i) t += wss[i];
        wss[0] = t;
    }
    __syncthreads();
    const float tot = wss[0];
    const float s10 = emb_scale[0] / fmaxf(sqrtf(tot), 1e-12f);

    embh[(size_t)bk * CCH + c] = (_Float16)(v * s10);
    if (threadIdx.x == 0) {
        sq[bk]     = tot * s10 * s10;          // ||scaled row||^2 (fp32 exact)
        negmin[bk] = __float_as_uint(BIGF);    // init hardest-negative buffer
    }
}

// ---------------------------------------------------------------------------
// Kernel 2: one 64x64 Gram tile per 128-thread block (4 waves, 2x2 WMMA tiles
// per wave). Async global->LDS staged f16 tiles, fragments assembled per the
// CDNA5 WMMA VGPR layouts, then distance + mask + shuffle row-min + atomics.
// ---------------------------------------------------------------------------
__global__ __launch_bounds__(128) void gram_min(
    const _Float16* __restrict__ embh, const float* __restrict__ sq,
    const int* __restrict__ reg_mask, unsigned* __restrict__ negmin)
{
    __shared__ _Float16 As[TILE * LDW];
    __shared__ _Float16 Bs[TILE * LDW];
    __shared__ float sqI[TILE], sqJ[TILE], penJ[TILE];

    const int b    = blockIdx.z;
    const int i0   = blockIdx.y * TILE;
    const int j0   = blockIdx.x * TILE;
    const int tid  = threadIdx.x;
    const int wave = tid >> 5;
    const int lane = tid & 31;
    const int hv   = lane >> 4;        // half-wave select
    const int m    = lane & 15;
    const int iw   = (wave >> 1) * 32; // wave's i offset in tile
    const int jw   = (wave & 1) * 32;  // wave's j offset in tile

    const size_t baseI = ((size_t)b * KPTS + i0) * CCH;
    const size_t baseJ = ((size_t)b * KPTS + j0) * CCH;

    if (tid < TILE) {
        sqI[tid] = sq[b * KPTS + i0 + tid];
    } else {
        const int t = tid - TILE;
        sqJ[t]  = sq[b * KPTS + j0 + t];
        penJ[t] = (reg_mask[b * KPTS + j0 + t] > 0) ? 0.f : 1.f;
    }

    v8f acc[2][2];
    #pragma unroll
    for (int ti = 0; ti < 2; ++ti)
        #pragma unroll
        for (int tj = 0; tj < 2; ++tj)
            #pragma unroll
            for (int e = 0; e < 8; ++e) acc[ti][tj][e] = 0.f;

    #pragma unroll
    for (int ch = 0; ch < CCH / CHUNK; ++ch) {
        __syncthreads();
        // stage 64 rows x 128 halves of A and B tiles
        #pragma unroll
        for (int u = 0; u < 8; ++u) {
            const int unit = tid + u * 128;       // 0..1023
            const int r  = unit >> 4;
            const int cv = unit & 15;
            const size_t goff = (size_t)r * CCH + ch * CHUNK + cv * 8;
            const int    loff = r * LDW + cv * 8;
#if HAVE_ASYNC_LDS
            __builtin_amdgcn_global_load_async_to_lds_b128(
                (ASG v4i_*)(void*)(embh + baseI + goff),
                (ASL v4i_*)(void*)&As[loff], 0, 0);
            __builtin_amdgcn_global_load_async_to_lds_b128(
                (ASG v4i_*)(void*)(embh + baseJ + goff),
                (ASL v4i_*)(void*)&Bs[loff], 0, 0);
#else
            uint4 da = *reinterpret_cast<const uint4*>(embh + baseI + goff);
            uint4 db = *reinterpret_cast<const uint4*>(embh + baseJ + goff);
            *reinterpret_cast<uint4*>(&As[loff]) = da;
            *reinterpret_cast<uint4*>(&Bs[loff]) = db;
#endif
        }
#if HAVE_ASYNC_LDS
        wait_async_zero();
#endif
        __syncthreads();

        #pragma unroll
        for (int kc = 0; kc < CHUNK / 32; ++kc) {
            FragH a[2], bb[2];
            // A 16x32 f16: lane<16 -> M=m, K={0..7,16..23}; lane>=16 -> K+8
            #pragma unroll
            for (int ti = 0; ti < 2; ++ti) {
                const int row  = iw + ti * 16 + m;
                const int off1 = row * LDW + kc * 32 + hv * 8;
                a[ti].q[0] = *reinterpret_cast<const uint4*>(&As[off1]);
                a[ti].q[1] = *reinterpret_cast<const uint4*>(&As[off1 + 16]);
            }
            // B 32x16: lane<16 -> N=m, K={0..7,16..23}; lane>=16 -> K+8
            #pragma unroll
            for (int tj = 0; tj < 2; ++tj) {
                const int row = jw + tj * 16 + m;
                const int off = row * LDW + kc * 32 + hv * 8;
                bb[tj].q[0] = *reinterpret_cast<const uint4*>(&Bs[off]);
                bb[tj].q[1] = *reinterpret_cast<const uint4*>(&Bs[off + 16]);
            }
            #pragma unroll
            for (int ti = 0; ti < 2; ++ti)
                #pragma unroll
                for (int tj = 0; tj < 2; ++tj)
                    acc[ti][tj] = __builtin_amdgcn_wmma_f32_16x16x32_f16(
                        false, a[ti].v, false, bb[tj].v,
                        (short)0, acc[ti][tj], false, false);
        }
    }

    // distances, masking, row-min, atomic hardest-negative update.
    // C/D layout: VGPR r -> M = r (lanes 0-15) / r+8 (lanes 16-31), N = lane&15.
    #pragma unroll
    for (int ti = 0; ti < 2; ++ti) {
        float rmin[8];
        #pragma unroll
        for (int r = 0; r < 8; ++r) rmin[r] = BIGF;
        #pragma unroll
        for (int tj = 0; tj < 2; ++tj) {
            const int jl  = jw + tj * 16 + m;
            const int j   = j0 + jl;
            const float sj  = sqJ[jl];
            const float pen = penJ[jl];
            #pragma unroll
            for (int r = 0; r < 8; ++r) {
                const int il = iw + ti * 16 + r + hv * 8;
                const int i  = i0 + il;
                float d2 = sqI[il] + sj - 2.f * acc[ti][tj][r];
                float d  = sqrtf(fmaxf(d2, 0.f));
                if (i == j || pen > 0.f) d = BIGF;   // exclude self + invalid j
                rmin[r] = fminf(rmin[r], d);
            }
        }
        #pragma unroll
        for (int r = 0; r < 8; ++r) {
            float v = rmin[r];
            v = fminf(v, __shfl_xor(v, 1, 32));
            v = fminf(v, __shfl_xor(v, 2, 32));
            v = fminf(v, __shfl_xor(v, 4, 32));
            v = fminf(v, __shfl_xor(v, 8, 32));
            if (m == 0) {
                const int il = iw + ti * 16 + r + hv * 8;
                atomicMin(&negmin[b * KPTS + i0 + il], __float_as_uint(v));
            }
        }
    }
}

// ---------------------------------------------------------------------------
// Kernel 3: hinge + masked mean over B*K anchors in a single 1024-thread block.
// ---------------------------------------------------------------------------
__global__ __launch_bounds__(1024) void final_reduce(
    const unsigned* __restrict__ negmin, const int* __restrict__ reg_mask,
    float* __restrict__ out)
{
    float sh = 0.f, sv = 0.f;
    for (int idx = threadIdx.x; idx < BATCH * KPTS; idx += 1024) {
        const float d  = __uint_as_float(negmin[idx]);
        const float h  = fmaxf(MARGINF - d, 0.f);
        const float vf = (reg_mask[idx] > 0) ? 1.f : 0.f;
        sh += h * vf;
        sv += vf;
    }
    #pragma unroll
    for (int s = 16; s >= 1; s >>= 1) {
        sh += __shfl_xor(sh, s, 32);
        sv += __shfl_xor(sv, s, 32);
    }
    __shared__ float ah[32], av[32];
    const int w = threadIdx.x >> 5, l = threadIdx.x & 31;
    if (l == 0) { ah[w] = sh; av[w] = sv; }
    __syncthreads();
    if (threadIdx.x == 0) {
        float th = 0.f, tv = 0.f;
        #pragma unroll
        for (int i = 0; i < 32; ++i) { th += ah[i]; tv += av[i]; }
        out[0] = th / tv;
    }
}

// ---------------------------------------------------------------------------
extern "C" void kernel_launch(void* const* d_in, const int* in_sizes, int n_in,
                              void* d_out, int out_size, void* d_ws, size_t ws_size,
                              hipStream_t stream)
{
    const float* out_id    = (const float*)d_in[0];   // [B,C,H,W] f32
    const int*   ind       = (const int*)d_in[1];     // [B,K]
    const int*   reg_mask  = (const int*)d_in[2];     // [B,K]
    const float* emb_scale = (const float*)d_in[3];   // scalar

    char* ws = (char*)d_ws;
    _Float16* embh   = (_Float16*)ws;                                   // 4 MB
    float*    sq     = (float*)(ws + (size_t)BATCH * KPTS * CCH * 2);   // 32 KB
    unsigned* negmin = (unsigned*)(ws + (size_t)BATCH * KPTS * CCH * 2
                                      + (size_t)BATCH * KPTS * 4);      // 32 KB

    gather_norm<<<BATCH * KPTS, 256, 0, stream>>>(out_id, ind, emb_scale,
                                                  embh, sq, negmin);

    dim3 grid(KPTS / TILE, KPTS / TILE, BATCH);     // 16 x 16 x 8
    gram_min<<<grid, 128, 0, stream>>>(embh, sq, reg_mask, negmin);

    final_reduce<<<1, 1024, 0, stream>>>(negmin, reg_mask, (float*)d_out);
}
